// DiMBlock_90452011254018
// MI455X (gfx1250) — matrix operand order repaired and checked by reading, not compiled, not run on hardware
//
#include <hip/hip_runtime.h>

// ---------------- problem dims (match reference) ----------------
#define BB   2
#define HS   32
#define WSZ  32
#define LL   (HS*WSZ)        // 1024
#define DIMC 256
#define DI   512             // d_inner
#define DSN  64              // d_state
#define DTR  16              // dt_rank
#define KD   4               // scan directions
#define HID  1024
#define NR   (DTR + 2*DSN)   // 144
#define BL   (BB*LL)         // 2048 rows

typedef __attribute__((ext_vector_type(16))) _Float16 v16h;
typedef __attribute__((ext_vector_type(8)))  _Float16 v8h;
typedef __attribute__((ext_vector_type(8)))  float    v8f;
typedef __attribute__((ext_vector_type(4)))  int      v4i;

#define AS1 __attribute__((address_space(1)))
#define AS3 __attribute__((address_space(3)))

// gfx1250 async global->LDS path (ASYNCcnt-tracked); fall back to plain LDS staging.
#if defined(__has_builtin)
#if __has_builtin(__builtin_amdgcn_global_load_async_to_lds_b128) && \
    __has_builtin(__builtin_amdgcn_s_wait_asynccnt)
#define USE_ASYNC_LDS 1
#endif
#endif
#ifndef USE_ASYNC_LDS
#define USE_ASYNC_LDS 0
#endif

__device__ __forceinline__ float silu_f(float x){ return x / (1.f + __expf(-x)); }

// ---------------- weight convert + transpose: src (K,N) f32 -> dst (N,K) f16 -------------
__global__ void f32_to_f16T(const float* __restrict__ src, _Float16* __restrict__ dst,
                            int K, int N) {
  long idx = (long)blockIdx.x * blockDim.x + threadIdx.x;
  if (idx >= (long)K * N) return;
  int k = (int)(idx / N), n = (int)(idx % N);
  dst[(long)n * K + k] = (_Float16)src[idx];
}

// ---------------- mod = silu(c) @ W_ada + b_ada ----------------
__global__ void mod_kernel(const float* __restrict__ c, const float* __restrict__ W,
                           const float* __restrict__ bias, float* __restrict__ mod) {
  int o = blockIdx.x * blockDim.x + threadIdx.x;   // 0..6*DIMC-1
  int b = blockIdx.y;
  if (o >= 6 * DIMC) return;
  float acc = bias[o];
  for (int i = 0; i < DIMC; ++i) {
    float cv = c[b * DIMC + i];
    acc = fmaf(silu_f(cv), W[(long)i * 6 * DIMC + o], acc);
  }
  mod[b * 6 * DIMC + o] = acc;
}

// ---------------- LN (no affine) + modulate -> f16, DIMC channels ----------------
__global__ void ln_mod_f16(const float* __restrict__ x, const float* __restrict__ mod,
                           int shOff, int scOff, _Float16* __restrict__ out) {
  int row = blockIdx.x;                 // b*LL + l
  int b   = row / LL;
  int t   = threadIdx.x;                // 256
  __shared__ float red[DIMC];
  float v = x[(long)row * DIMC + t];
  red[t] = v; __syncthreads();
  for (int s = DIMC/2; s > 0; s >>= 1) { if (t < s) red[t] += red[t + s]; __syncthreads(); }
  float mu = red[0] * (1.f / DIMC); __syncthreads();
  float xc = v - mu;
  red[t] = xc * xc; __syncthreads();
  for (int s = DIMC/2; s > 0; s >>= 1) { if (t < s) red[t] += red[t + s]; __syncthreads(); }
  float rstd = rsqrtf(red[0] * (1.f / DIMC) + 1e-6f);
  float sh = mod[b * 6 * DIMC + shOff + t];
  float sc = mod[b * 6 * DIMC + scOff + t];
  out[(long)row * DIMC + t] = (_Float16)(xc * rstd * (1.f + sc) + sh);
}

// ---------------- WMMA GEMM: C(M,N) = A(M,Kd) f16 @ BT(N,Kd)^T f16 (+bias, act) ----------
// One wave = 1 M-tile x NT N-tiles (A-fragment reused across NT WMMAs per K-step).
// NT is compile-time; caller guarantees (N/16) % NT == 0 -> no guards, no divergence,
// accumulators stay in VGPRs, EXEC stays all-ones (WMMA requirement).
// 8 waves / 256-thread block; batched via blockIdx.y.
// ACT: 0=none, 2=gelu(tanh). HALF_OUT: f16 store.
template<int ACT, bool HALF_OUT, int NT>
__global__ void wmma_gemm(const _Float16* __restrict__ A, const _Float16* __restrict__ BT,
                          void* __restrict__ Cout, const float* __restrict__ bias,
                          int M, int N, int Kd, long sA, long sB, long sC) {
  int batch = blockIdx.y;
  const _Float16* Ab  = A  + (long)batch * sA;
  const _Float16* BTb = BT + (long)batch * sB;
  int lane = threadIdx.x & 31;
  int wave = threadIdx.x >> 5;
  int tiles_n  = N >> 4;
  int groups_n = tiles_n / NT;
  int group = blockIdx.x * 8 + wave;
  if (group >= (M >> 4) * groups_n) return;       // wave-uniform
  int tm = group / groups_n, tg = group % groups_n;
  int l16   = lane & 15;
  int khalf = lane >> 4;
  const _Float16* arow  = Ab  + (long)((tm << 4) + l16) * Kd + khalf * 8;
  const _Float16* brow0 = BTb + ((long)(tg * NT * 16 + l16)) * Kd + khalf * 8;
  v8f acc[NT];
  #pragma unroll
  for (int j = 0; j < NT; ++j) acc[j] = (v8f){};
  for (int k0 = 0; k0 < Kd; k0 += 32) {
    v8h alo = *(const v8h*)(arow + k0);            // A[m, k0 + khalf*8 .. +7]
    v8h ahi = *(const v8h*)(arow + k0 + 16);       // A[m, k0+16+khalf*8 .. +7]
    v16h af = __builtin_shufflevector(alo, ahi, 0,1,2,3,4,5,6,7,8,9,10,11,12,13,14,15);
    #pragma unroll
    for (int j = 0; j < NT; ++j) {
      v8h blo = *(const v8h*)(brow0 + ((long)j << 4) * Kd + k0);
      v8h bhi = *(const v8h*)(brow0 + ((long)j << 4) * Kd + k0 + 16);
      v16h bf = __builtin_shufflevector(blo, bhi, 0,1,2,3,4,5,6,7,8,9,10,11,12,13,14,15);
      acc[j] = __builtin_amdgcn_wmma_f32_16x16x32_f16(false, af, false, bf,
                                                      (short)0, acc[j], false, false);
    }
  }
  int rbase = (tm << 4) + khalf * 8;
  #pragma unroll
  for (int j = 0; j < NT; ++j) {
    int col = ((tg * NT + j) << 4) + l16;
    float bi = bias ? bias[col] : 0.f;
    #pragma unroll
    for (int v = 0; v < 8; ++v) {
      float val = acc[j][v] + bi;
      if (ACT == 2) {
        float u = 0.7978845608028654f * (val + 0.044715f * val * val * val);
        val = 0.5f * val * (1.f + tanhf(u));
      }
      long off = batch * sC + (long)(rbase + v) * N + col;
      if (HALF_OUT) ((_Float16*)Cout)[off] = (_Float16)val;
      else          ((float*)Cout)[off] = val;
    }
  }
}

// ---------------- depthwise 3x3 SAME conv + SiLU, scatter 4 directions -> xs16 ------------
__global__ void conv_dirs(const float* __restrict__ xz, const float* __restrict__ conv_w,
                          const float* __restrict__ conv_b, _Float16* __restrict__ xs16) {
  long idx = (long)blockIdx.x * blockDim.x + threadIdx.x;
  if (idx >= (long)BB * LL * DI) return;
  int ch = (int)(idx % DI);
  long r = idx / DI;
  int w = (int)(r % WSZ); r /= WSZ;
  int h = (int)(r % HS);
  int b = (int)(r / HS);
  float acc = conv_b[ch];
  for (int kh = -1; kh <= 1; ++kh) {
    int hh = h + kh; if (hh < 0 || hh >= HS) continue;
    for (int kw = -1; kw <= 1; ++kw) {
      int ww = w + kw; if (ww < 0 || ww >= WSZ) continue;
      float xv = xz[((long)b * LL + hh * WSZ + ww) * (2 * DI) + ch];   // xi = xz[:, :DI]
      acc = fmaf(xv, conv_w[((kh + 1) * 3 + (kw + 1)) * DI + ch], acc);
    }
  }
  _Float16 hv = (_Float16)silu_f(acc);
  int lr = h * WSZ + w;        // row-major index
  int lc = w * HS + h;         // col-major index
  long base = (long)b * LL;
  const long S = (long)BB * LL * DI;
  xs16[0 * S + (base + lr) * DI + ch]            = hv;
  xs16[1 * S + (base + lc) * DI + ch]            = hv;
  xs16[2 * S + (base + (LL - 1 - lr)) * DI + ch] = hv;
  xs16[3 * S + (base + (LL - 1 - lc)) * DI + ch] = hv;
}

// ---------------- dt = softplus(dt_r @ W_dt + dt_bias), (K, BL, DI) ----------------
__global__ void dt_kernel(const float* __restrict__ xdbl, const float* __restrict__ W_dt,
                          const float* __restrict__ dt_bias, float* __restrict__ dtv) {
  long idx = (long)blockIdx.x * blockDim.x + threadIdx.x;
  if (idx >= (long)KD * BL * DI) return;
  int d = (int)(idx % DI);
  long r = idx / DI;
  int bl = (int)(r % BL);
  int k  = (int)(r / BL);
  float accv = dt_bias[k * DI + d];
  const float* xr = xdbl + ((long)k * BL + bl) * NR;
  const float* wd = W_dt + (long)k * DTR * DI + d;
  #pragma unroll
  for (int j = 0; j < DTR; ++j) accv = fmaf(xr[j], wd[(long)j * DI], accv);
  dtv[idx] = (accv > 20.f) ? accv : log1pf(__expf(accv));
}

// ---------------- selective scan: 4 threads/channel, 16 states each -------------------
// B/C (one contiguous 512B chunk per step) double-buffered in LDS; prefetched with
// GLOBAL_LOAD_ASYNC_TO_LDS_B128 (ASYNCcnt) when available.
__global__ void scan_kernel(const _Float16* __restrict__ xs16, const float* __restrict__ dtv,
                            const float* __restrict__ xdbl, const float* __restrict__ A_log,
                            const float* __restrict__ Dp, float* __restrict__ ys) {
  int blk  = blockIdx.x;                      // BB*KD*(DI/128) = 32 blocks of 512
  int dblk = blk % (DI / 128); blk /= (DI / 128);
  int k    = blk % KD;
  int b    = blk / KD;
  int tid  = threadIdx.x;                     // 512
  int d    = dblk * 128 + (tid >> 2);
  int sq   = (tid & 3) * 16;
  float Ar[16], hreg[16];
  #pragma unroll
  for (int s = 0; s < 16; ++s) {
    Ar[s]   = -__expf(A_log[((long)k * DI + d) * DSN + sq + s]);
    hreg[s] = 0.f;
  }
  float Dval = Dp[k * DI + d];
  __shared__ __align__(16) float BC[2][2 * DSN];     // [buf][0..63]=B, [64..127]=C
  const long S = (long)BB * LL * DI;
  long rowbase = (long)b * LL;
  const float* bcbase = xdbl + ((long)k * BL + rowbase) * NR + DTR;

  // stage t = 0 into buffer 0
#if USE_ASYNC_LDS
  if (tid < 32) {
    __builtin_amdgcn_global_load_async_to_lds_b128(
        (AS1 v4i*)(bcbase + tid * 4), (AS3 v4i*)&BC[0][tid * 4], 0, 0);
    __builtin_amdgcn_s_wait_asynccnt(0);
  }
#else
  if (tid < 2 * DSN) BC[0][tid] = bcbase[tid];
#endif
  __syncthreads();

  for (int t = 0; t < LL; ++t) {
    int buf = t & 1;
    // prefetch t+1 into the other buffer (its readers passed the previous barrier)
    if (t + 1 < LL) {
#if USE_ASYNC_LDS
      if (tid < 32)
        __builtin_amdgcn_global_load_async_to_lds_b128(
            (AS1 v4i*)(bcbase + (long)(t + 1) * NR + tid * 4),
            (AS3 v4i*)&BC[buf ^ 1][tid * 4], 0, 0);
#else
      if (tid < 2 * DSN) BC[buf ^ 1][tid] = bcbase[(long)(t + 1) * NR + tid];
#endif
    }
    long bl = rowbase + t;
    float u   = (float)xs16[(long)k * S + bl * DI + d];
    float dtq = dtv[(long)k * S + bl * DI + d];
    float du  = dtq * u;
    float acc = 0.f;
    const float* Bs = &BC[buf][0];
    const float* Cs = &BC[buf][DSN];
    #pragma unroll
    for (int s = 0; s < 16; ++s) {
      hreg[s] = fmaf(hreg[s], __expf(dtq * Ar[s]), du * Bs[sq + s]);
      acc     = fmaf(hreg[s], Cs[sq + s], acc);
    }
    acc += __shfl_xor(acc, 1, 32);              // reduce the 4 state-quarters
    acc += __shfl_xor(acc, 2, 32);
    if ((tid & 3) == 0)
      ys[(long)k * S + bl * DI + d] = acc + u * Dval;
#if USE_ASYNC_LDS
    if (tid < 32) __builtin_amdgcn_s_wait_asynccnt(0);
#endif
    __syncthreads();
  }
}

// ---------------- merge 4 directions + LN*w+b + silu(z) gate -> f16 ----------------
__global__ void merge_ln_gate(const float* __restrict__ ys, const float* __restrict__ xz,
                              const float* __restrict__ ln_w, const float* __restrict__ ln_b,
                              _Float16* __restrict__ yg16) {
  int row = blockIdx.x;                       // b*LL + l
  int b = row / LL, l = row % LL;
  int h = l / WSZ, w = l % WSZ;
  int lc = w * HS + h;
  int t = threadIdx.x;                        // 512 = DI
  const long S = (long)BB * LL * DI;
  long base = (long)b * LL;
  float y = ys[0 * S + (base + l) * DI + t]
          + ys[1 * S + (base + lc) * DI + t]               // uncol(ys1)
          + ys[2 * S + (base + (LL - 1 - l)) * DI + t]     // rev(ys2)
          + ys[3 * S + (base + (LL - 1 - lc)) * DI + t];   // uncol(rev(ys3))
  __shared__ float red[DI];
  red[t] = y; __syncthreads();
  for (int s = DI/2; s > 0; s >>= 1) { if (t < s) red[t] += red[t + s]; __syncthreads(); }
  float mu = red[0] * (1.f / DI); __syncthreads();
  float xc = y - mu;
  red[t] = xc * xc; __syncthreads();
  for (int s = DI/2; s > 0; s >>= 1) { if (t < s) red[t] += red[t + s]; __syncthreads(); }
  float rstd = rsqrtf(red[0] * (1.f / DI) + 1e-6f);
  float v = xc * rstd * ln_w[t] + ln_b[t];
  float z = xz[(long)row * (2 * DI) + DI + t];
  yg16[(long)row * DI + t] = (_Float16)(v * silu_f(z));
}

// ---------------- residual: out = x + gate[b] * y ----------------
__global__ void residual_kernel(const float* __restrict__ x, const float* __restrict__ y,
                                const float* __restrict__ mod, int gOff,
                                float* __restrict__ out) {
  long idx = (long)blockIdx.x * blockDim.x + threadIdx.x;
  if (idx >= (long)BL * DIMC) return;
  int ch = (int)(idx % DIMC);
  int b  = (int)((idx / DIMC) / LL);
  out[idx] = x[idx] + mod[b * 6 * DIMC + gOff + ch] * y[idx];
}

// ============================= host launcher =============================
extern "C" void kernel_launch(void* const* d_in, const int* in_sizes, int n_in,
                              void* d_out, int out_size, void* d_ws, size_t ws_size,
                              hipStream_t stream) {
  (void)in_sizes; (void)n_in; (void)out_size; (void)ws_size;
  const float* x       = (const float*)d_in[0];
  const float* c       = (const float*)d_in[1];
  const float* W_ada   = (const float*)d_in[2];
  const float* b_ada   = (const float*)d_in[3];
  const float* W_in    = (const float*)d_in[4];
  const float* b_in    = (const float*)d_in[5];
  const float* conv_w  = (const float*)d_in[6];
  const float* conv_b  = (const float*)d_in[7];
  const float* W_xproj = (const float*)d_in[8];
  const float* W_dt    = (const float*)d_in[9];
  const float* dt_bias = (const float*)d_in[10];
  const float* A_log   = (const float*)d_in[11];
  const float* Dp      = (const float*)d_in[12];
  const float* ln_w    = (const float*)d_in[13];
  const float* ln_b    = (const float*)d_in[14];
  const float* W_out   = (const float*)d_in[15];
  const float* b_out   = (const float*)d_in[16];
  const float* W_fc1   = (const float*)d_in[17];
  const float* b_fc1   = (const float*)d_in[18];
  const float* W_fc2   = (const float*)d_in[19];
  const float* b_fc2   = (const float*)d_in[20];
  float* out = (float*)d_out;

  char* p = (char*)d_ws;
  auto alloc = [&](size_t bytes) -> void* {
    void* r = (void*)p; p += (bytes + 255) & ~(size_t)255; return r;
  };
  float*    mod   = (float*)   alloc((size_t)BB * 6 * DIMC * 4);
  _Float16* h16   = (_Float16*)alloc((size_t)BL * DIMC * 2);
  _Float16* WinT  = (_Float16*)alloc((size_t)2 * DI * DIMC * 2);   // (1024,256)
  _Float16* WxT   = (_Float16*)alloc((size_t)KD * NR * DI * 2);    // (4,144,512)
  _Float16* WoutT = (_Float16*)alloc((size_t)DIMC * DI * 2);       // (256,512)
  _Float16* Wfc1T = (_Float16*)alloc((size_t)HID * DIMC * 2);      // (1024,256)
  _Float16* Wfc2T = (_Float16*)alloc((size_t)DIMC * HID * 2);      // (256,1024)
  float*    xz    = (float*)   alloc((size_t)BL * 2 * DI * 4);
  _Float16* xs16  = (_Float16*)alloc((size_t)KD * BL * DI * 2);
  float*    xdbl  = (float*)   alloc((size_t)KD * BL * NR * 4);
  float*    dtv   = (float*)   alloc((size_t)KD * BL * DI * 4);
  float*    ys    = (float*)   alloc((size_t)KD * BL * DI * 4);
  _Float16* yg16  = (_Float16*)alloc((size_t)BL * DI * 2);
  float*    hy    = (float*)   alloc((size_t)BL * DIMC * 4);
  float*    x1    = (float*)   alloc((size_t)BL * DIMC * 4);
  _Float16* m16   = (_Float16*)alloc((size_t)BL * DIMC * 2);
  _Float16* g116  = (_Float16*)alloc((size_t)BL * HID * 2);
  float*    mlp   = (float*)   alloc((size_t)BL * DIMC * 4);

  auto cgrid = [](long n) { return (int)((n + 255) / 256); };
  auto ggrid = [](int M, int N, int nt) {               // wave-groups for wmma_gemm
    int groups = (M / 16) * ((N / 16) / nt);
    return (groups + 7) / 8;
  };

  // weights -> f16 transposed
  f32_to_f16T<<<cgrid((long)DIMC * 2 * DI), 256, 0, stream>>>(W_in, WinT, DIMC, 2 * DI);
  for (int k = 0; k < KD; ++k)
    f32_to_f16T<<<cgrid((long)DI * NR), 256, 0, stream>>>(
        W_xproj + (size_t)k * DI * NR, WxT + (size_t)k * NR * DI, DI, NR);
  f32_to_f16T<<<cgrid((long)DI * DIMC), 256, 0, stream>>>(W_out, WoutT, DI, DIMC);
  f32_to_f16T<<<cgrid((long)DIMC * HID), 256, 0, stream>>>(W_fc1, Wfc1T, DIMC, HID);
  f32_to_f16T<<<cgrid((long)HID * DIMC), 256, 0, stream>>>(W_fc2, Wfc2T, HID, DIMC);

  // adaLN modulation
  mod_kernel<<<dim3(6, BB), 256, 0, stream>>>(c, W_ada, b_ada, mod);

  // LN + modulate (msa) -> h16
  ln_mod_f16<<<BL, DIMC, 0, stream>>>(x, mod, 0, DIMC, h16);

  // xz = h16 @ W_in + b_in  (2048 x 1024 x 256), 64 N-tiles -> NT=4
  wmma_gemm<0, false, 4><<<dim3(ggrid(BL, 2 * DI, 4), 1), 256, 0, stream>>>(
      h16, WinT, xz, b_in, BL, 2 * DI, DIMC, 0, 0, 0);

  // depthwise conv + silu + 4-direction scatter
  conv_dirs<<<cgrid((long)BB * LL * DI), 256, 0, stream>>>(xz, conv_w, conv_b, xs16);

  // x_dbl[k] = xs16[k] @ W_xproj[k]  (batched over K; 2048 x 144 x 512), 9 N-tiles -> NT=3
  wmma_gemm<0, false, 3><<<dim3(ggrid(BL, NR, 3), KD), 256, 0, stream>>>(
      xs16, WxT, xdbl, nullptr, BL, NR, DI,
      (long)BL * DI, (long)NR * DI, (long)BL * NR);

  // dt projection + softplus
  dt_kernel<<<cgrid((long)KD * BL * DI), 256, 0, stream>>>(xdbl, W_dt, dt_bias, dtv);

  // selective scan
  scan_kernel<<<BB * KD * (DI / 128), 512, 0, stream>>>(xs16, dtv, xdbl, A_log, Dp, ys);

  // merge directions + LN + silu(z) gate -> yg16
  merge_ln_gate<<<BL, DI, 0, stream>>>(ys, xz, ln_w, ln_b, yg16);

  // hy = yg16 @ W_out + b_out  (2048 x 256 x 512), 16 N-tiles -> NT=4
  wmma_gemm<0, false, 4><<<dim3(ggrid(BL, DIMC, 4), 1), 256, 0, stream>>>(
      yg16, WoutT, hy, b_out, BL, DIMC, DI, 0, 0, 0);

  // x1 = x + g_msa * hy
  residual_kernel<<<cgrid((long)BL * DIMC), 256, 0, stream>>>(x, hy, mod, 2 * DIMC, x1);

  // LN + modulate (mlp) -> m16
  ln_mod_f16<<<BL, DIMC, 0, stream>>>(x1, mod, 3 * DIMC, 4 * DIMC, m16);

  // g1 = gelu(m16 @ W_fc1 + b_fc1) -> f16  (2048 x 1024 x 256), NT=4
  wmma_gemm<2, true, 4><<<dim3(ggrid(BL, HID, 4), 1), 256, 0, stream>>>(
      m16, Wfc1T, g116, b_fc1, BL, HID, DIMC, 0, 0, 0);

  // mlp = g1 @ W_fc2 + b_fc2  (2048 x 256 x 1024), NT=4
  wmma_gemm<0, false, 4><<<dim3(ggrid(BL, DIMC, 4), 1), 256, 0, stream>>>(
      g116, Wfc2T, mlp, b_fc2, BL, DIMC, HID, 0, 0, 0);

  // out = x1 + g_mlp * mlp
  residual_kernel<<<cgrid((long)BL * DIMC), 256, 0, stream>>>(x1, mlp, mod, 5 * DIMC, out);
}
